// RelationAwareMHA_17128329576858
// MI455X (gfx1250) — compile-verified, matrix operand order checked by reading, not run on hardware
//
#include <hip/hip_runtime.h>
#include <hip/hip_bf16.h>

// Problem constants (match reference)
#define B_  2
#define L_  2048
#define D_  1024
#define H_  16
#define DK_ 64
#define NREL_ 64

typedef __attribute__((ext_vector_type(16))) __bf16 v16bf;
typedef __attribute__((ext_vector_type(8)))  float  v8f;

union BT { unsigned u[8]; v16bf v; };

__device__ __forceinline__ v8f wmma_bf16(const v16bf& a, const v16bf& b, v8f c) {
  return __builtin_amdgcn_wmma_f32_16x16x32_bf16(false, a, false, b, (short)0, c, false, false);
}

// ---------------- elementwise converts ----------------
__global__ void f32_to_bf16_kernel(const float* __restrict__ in, __bf16* __restrict__ out, int n) {
  int idx = blockIdx.x * 256 + threadIdx.x;
  if (idx < n) out[idx] = (__bf16)in[idx];
}

// Wt[n*1024 + k] = (bf16) W[k*1024 + n]   (column-major for WMMA B-tiles)
__global__ void transpose_w_kernel(const float* __restrict__ W, __bf16* __restrict__ Wt) {
  int idx = blockIdx.x * 256 + threadIdx.x;           // 0 .. 1M-1
  int n = idx >> 10, k = idx & 1023;
  Wt[idx] = (__bf16)W[k * 1024 + n];
}

// ---------------- blocked GEMM: 16 rows x 64 cols per wave, 4 waves/block ----
// A: 4096x1024 bf16 row-major.  Bt: col-major bf16 (Bt[n*1024+k]).
// MODE 0: bf16 out (B,H,L,64)   (Q, K)
// MODE 1: bf16 out (B,H,64,L)   (V transposed)
// MODE 2: f32  out row-major 4096x1024 (final projection)
template <int MODE>
__global__ __launch_bounds__(128) void gemm16x64_kernel(const __bf16* __restrict__ Am,
                                                        const __bf16* __restrict__ Bt,
                                                        void* __restrict__ outp) {
  int wave  = threadIdx.x >> 5;
  int lane  = threadIdx.x & 31;
  int tile  = blockIdx.x * 4 + wave;   // 0 .. 4095
  int tg    = tile & 15;               // 16 col groups of 64
  int tm    = tile >> 4;               // 256 row tiles of 16
  int mrow  = lane & 15;
  int khalf = lane >> 4;

  int row = tm * 16 + mrow;

  v8f c0 = {}, c1 = {}, c2 = {}, c3 = {};
  for (int k0 = 0; k0 < 1024; k0 += 32) {
    BT a;
#pragma unroll
    for (int v = 0; v < 8; ++v) {
      int kk = k0 + ((v >> 2) * 16) + khalf * 8 + ((v & 3) * 2);
      a.u[v] = *(const unsigned*)(Am + (size_t)row * 1024 + kk);
    }
    v8f* cc[4] = { &c0, &c1, &c2, &c3 };
#pragma unroll
    for (int n = 0; n < 4; ++n) {
      BT b;
      int col = tg * 64 + n * 16 + mrow;
#pragma unroll
      for (int v = 0; v < 8; ++v) {
        int kk = k0 + ((v >> 2) * 16) + khalf * 8 + ((v & 3) * 2);
        b.u[v] = *(const unsigned*)(Bt + (size_t)col * 1024 + kk);
      }
      *cc[n] = wmma_bf16(a.v, b.v, *cc[n]);
    }
  }

  v8f cres[4] = { c0, c1, c2, c3 };
#pragma unroll
  for (int n = 0; n < 4; ++n) {
    int ncol = tg * 64 + n * 16 + (lane & 15);
#pragma unroll
    for (int r = 0; r < 8; ++r) {
      int grow = tm * 16 + r + 8 * khalf;            // 0..4095
      float fv = cres[n][r];
      if (MODE == 2) {
        ((float*)outp)[(size_t)grow * 1024 + ncol] = fv;
      } else {
        int bidx = grow >> 11;
        int i    = grow & (L_ - 1);
        int h    = ncol >> 6, d = ncol & 63;
        __bf16 val = (__bf16)fv;
        if (MODE == 0)
          ((__bf16*)outp)[(((size_t)(bidx * H_ + h) * L_ + i) * DK_) + d] = val;
        else
          ((__bf16*)outp)[(((size_t)(bidx * H_ + h) * DK_ + d) * L_) + i] = val;
      }
    }
  }
}

// ---------------- flash attention with relation bias (4 waves/block) --------
__global__ __launch_bounds__(128) void flash_kernel(const __bf16* __restrict__ q,
                                                    const __bf16* __restrict__ kmat,
                                                    const __bf16* __restrict__ vt,
                                                    const unsigned char* __restrict__ mask,
                                                    const int* __restrict__ rids,
                                                    const float* __restrict__ rel_emb,
                                                    __bf16* __restrict__ ctx) {
  int wave  = threadIdx.x >> 5;
  int lane  = threadIdx.x & 31;
  int id    = blockIdx.x * 4 + wave;  // 0 .. 4095
  int itile = id & 127;               // L/16
  int h     = (id >> 7) & (H_ - 1);
  int b     = id >> 11;
  int mrow  = lane & 15;
  int khalf = lane >> 4;

  __shared__ float relE[4][NREL_];
  __shared__ __align__(16) __bf16 Pst[4][16 * 32];

  relE[wave][lane]      = rel_emb[lane * H_ + h];
  relE[wave][lane + 32] = rel_emb[(lane + 32) * H_ + h];
  __threadfence_block();

  int i0 = itile * 16;
  const __bf16* qbase = q    + (size_t)(b * H_ + h) * L_ * DK_;
  const __bf16* kbase = kmat + (size_t)(b * H_ + h) * L_ * DK_;
  const __bf16* vbase = vt   + (size_t)(b * H_ + h) * DK_ * L_;

  // Q tiles in A-layout (16x32 each; d 0..31 and 32..63)
  BT qa[2];
#pragma unroll
  for (int t = 0; t < 2; ++t)
#pragma unroll
    for (int v = 0; v < 8; ++v) {
      int d = t * 32 + ((v >> 2) * 16) + khalf * 8 + ((v & 3) * 2);
      qa[t].u[v] = *(const unsigned*)(qbase + (size_t)(i0 + mrow) * DK_ + d);
    }

  v8f o0 = {}, o1 = {}, o2 = {}, o3 = {};
  float m[8], lsum[8];
#pragma unroll
  for (int r = 0; r < 8; ++r) { m[r] = -1e30f; lsum[r] = 0.0f; }

  for (int j0 = 0; j0 < L_; j0 += 32) {
    // ---- scores S = Q K^T for 32 columns (two 16x16 C-tiles)
    v8f s[2];
#pragma unroll
    for (int t = 0; t < 2; ++t) {
      BT blo, bhi;
      int j = j0 + t * 16 + mrow;
#pragma unroll
      for (int v = 0; v < 8; ++v) {
        int dlo = ((v >> 2) * 16) + khalf * 8 + ((v & 3) * 2);
        blo.u[v] = *(const unsigned*)(kbase + (size_t)j * DK_ + dlo);
        bhi.u[v] = *(const unsigned*)(kbase + (size_t)j * DK_ + dlo + 32);
      }
      v8f z = {};
      z = wmma_bf16(qa[0].v, blo.v, z);
      z = wmma_bf16(qa[1].v, bhi.v, z);
      s[t] = z;
    }

    // ---- scale + relation bias + mask
#pragma unroll
    for (int t = 0; t < 2; ++t) {
      int j = j0 + t * 16 + mrow;
#pragma unroll
      for (int r = 0; r < 8; ++r) {
        int i = i0 + r + 8 * khalf;
        size_t off = ((size_t)(b * L_ + i)) * L_ + j;
        float sv = s[t][r] * 0.125f + relE[wave][rids[off]];
        if (!mask[off]) sv = -1e30f;
        s[t][r] = sv;
      }
    }

    // ---- online softmax update (rows split over the two 16-lane halves)
    float sc[8];
#pragma unroll
    for (int r = 0; r < 8; ++r) {
      float v = fmaxf(s[0][r], s[1][r]);
#pragma unroll
      for (int msk = 8; msk >= 1; msk >>= 1)
        v = fmaxf(v, __shfl_xor(v, msk, 32));
      float mnew = fmaxf(m[r], v);
      sc[r] = __expf(m[r] - mnew);
      m[r] = mnew;
      lsum[r] *= sc[r];
    }
#pragma unroll
    for (int r = 0; r < 8; ++r) { o0[r] *= sc[r]; o1[r] *= sc[r]; o2[r] *= sc[r]; o3[r] *= sc[r]; }

#pragma unroll
    for (int r = 0; r < 8; ++r) {
      float p0 = __expf(s[0][r] - m[r]);
      float p1 = __expf(s[1][r] - m[r]);
      s[0][r] = p0; s[1][r] = p1;
      float ss = p0 + p1;
#pragma unroll
      for (int msk = 8; msk >= 1; msk >>= 1)
        ss += __shfl_xor(ss, msk, 32);
      lsum[r] += ss;
    }

    // ---- restage P (16x32) through this wave's LDS slice into A-layout
#pragma unroll
    for (int t = 0; t < 2; ++t)
#pragma unroll
      for (int r = 0; r < 8; ++r)
        Pst[wave][(r + 8 * khalf) * 32 + t * 16 + mrow] = (__bf16)s[t][r];
    __threadfence_block();

    BT pa;
#pragma unroll
    for (int v = 0; v < 8; ++v) {
      int pk = ((v >> 2) * 16) + khalf * 8 + ((v & 3) * 2);
      pa.u[v] = *(const unsigned*)(&Pst[wave][mrow * 32 + pk]);
    }
    __threadfence_block();

    // ---- O += P @ V  (4 d-tiles of 16; V stored transposed -> 32-bit loads)
    v8f* oo[4] = { &o0, &o1, &o2, &o3 };
#pragma unroll
    for (int dt = 0; dt < 4; ++dt) {
      BT bv;
      int d = dt * 16 + mrow;
#pragma unroll
      for (int v = 0; v < 8; ++v) {
        int j = j0 + ((v >> 2) * 16) + khalf * 8 + ((v & 3) * 2);
        bv.u[v] = *(const unsigned*)(vbase + (size_t)d * L_ + j);
      }
      *oo[dt] = wmma_bf16(pa.v, bv.v, *oo[dt]);
    }
  }

  // ---- normalize and write ctx (B, L, D) bf16 at columns h*64 + d
  v8f oarr[4] = { o0, o1, o2, o3 };
#pragma unroll
  for (int dt = 0; dt < 4; ++dt)
#pragma unroll
    for (int r = 0; r < 8; ++r) {
      int i = i0 + r + 8 * khalf;
      int d = dt * 16 + mrow;
      float val = oarr[dt][r] / lsum[r];
      ctx[((size_t)(b * L_ + i)) * D_ + h * DK_ + d] = (__bf16)val;
    }
}

extern "C" void kernel_launch(void* const* d_in, const int* in_sizes, int n_in,
                              void* d_out, int out_size, void* d_ws, size_t ws_size,
                              hipStream_t stream) {
  const float*         x    = (const float*)d_in[0];
  const unsigned char* mask = (const unsigned char*)d_in[1];
  const int*           rids = (const int*)d_in[2];
  const float*         Wq   = (const float*)d_in[3];
  const float*         Wk   = (const float*)d_in[4];
  const float*         Wv   = (const float*)d_in[5];
  const float*         Wo   = (const float*)d_in[6];
  const float*         rel  = (const float*)d_in[7];
  float* out = (float*)d_out;

  char* ws = (char*)d_ws;
  size_t off = 0;
  const size_t NX = (size_t)B_ * L_ * D_;          // 4,194,304
  const size_t NW = (size_t)D_ * D_;               // 1,048,576
  const size_t NQ = (size_t)B_ * H_ * L_ * DK_;    // 4,194,304

  __bf16* xh  = (__bf16*)(ws + off); off += NX * 2;
  __bf16* Wqt = (__bf16*)(ws + off); off += NW * 2;
  __bf16* Wkt = (__bf16*)(ws + off); off += NW * 2;
  __bf16* Wvt = (__bf16*)(ws + off); off += NW * 2;
  __bf16* Wot = (__bf16*)(ws + off); off += NW * 2;
  __bf16* qb  = (__bf16*)(ws + off); off += NQ * 2;
  __bf16* kb  = (__bf16*)(ws + off); off += NQ * 2;
  __bf16* vtb = (__bf16*)(ws + off); off += NQ * 2;
  __bf16* ctx = (__bf16*)(ws + off); off += NX * 2;   // 48 MiB total

  f32_to_bf16_kernel<<<(int)((NX + 255) / 256), 256, 0, stream>>>(x, xh, (int)NX);
  transpose_w_kernel<<<(int)(NW / 256), 256, 0, stream>>>(Wq, Wqt);
  transpose_w_kernel<<<(int)(NW / 256), 256, 0, stream>>>(Wk, Wkt);
  transpose_w_kernel<<<(int)(NW / 256), 256, 0, stream>>>(Wv, Wvt);
  transpose_w_kernel<<<(int)(NW / 256), 256, 0, stream>>>(Wo, Wot);

  // 4096 (16x64) tiles, 4 waves per block -> 1024 blocks
  gemm16x64_kernel<0><<<1024, 128, 0, stream>>>(xh, Wqt, qb);
  gemm16x64_kernel<0><<<1024, 128, 0, stream>>>(xh, Wkt, kb);
  gemm16x64_kernel<1><<<1024, 128, 0, stream>>>(xh, Wvt, vtb);

  // 4096 row-block/head units, 4 waves per block -> 1024 blocks
  flash_kernel<<<1024, 128, 0, stream>>>(qb, kb, vtb, mask, rids, rel, ctx);

  gemm16x64_kernel<2><<<1024, 128, 0, stream>>>(ctx, Wot, out);
}
